// Fast_Attention_8641474200015
// MI455X (gfx1250) — matrix operand order repaired and checked by reading, not compiled, not run on hardware
//
#include <hip/hip_runtime.h>
#include <hip/hip_fp16.h>

typedef __attribute__((ext_vector_type(16))) _Float16 v16h;
typedef __attribute__((ext_vector_type(8)))  float    v8f;
typedef __attribute__((ext_vector_type(4)))  unsigned int v4u;
typedef __attribute__((ext_vector_type(8)))  int      v8i;
typedef __attribute__((ext_vector_type(4)))  int      v4i;

#define LANE (threadIdx.x & 31)
#define WAVE (threadIdx.x >> 5)

constexpr int Bc  = 2;
constexpr int Nn  = 2048;
constexpr int DIMc = 512;
constexpr int Hc  = 8;
constexpr int DHc = 64;
constexpr int CATC = 2 * Hc * DHc;   // 1024
constexpr int CH  = 128;             // attention key-chunk staged in LDS via TDM

__device__ inline v8f wmma_f16(v16h a, v16h b, v8f c) {
    return __builtin_amdgcn_wmma_f32_16x16x32_f16(false, a, false, b, (short)0, c,
                                                  false, false);
}

// ---- Fragment loaders, contiguous k order K = 16*(lane/16) + j (wave32) ----
// A 16x32 f16: lane m = lane&15; kb = (lane>=16)?16:0; elem j -> K=kb+j
__device__ inline v16h load_a_f16(const _Float16* A, int lda, int lane) {
    int m = lane & 15; int kb = (lane & 16) ? 16 : 0;
    const _Float16* r = A + m * lda + kb;
    v16h a;
#pragma unroll
    for (int j = 0; j < 8; ++j) { a[j] = r[j]; a[8 + j] = r[8 + j]; }
    return a;
}
__device__ inline v16h load_a_f32(const float* A, int lda, int lane) {
    int m = lane & 15; int kb = (lane & 16) ? 16 : 0;
    const float* r = A + m * lda + kb;
    v16h a;
#pragma unroll
    for (int j = 0; j < 8; ++j) { a[j] = (_Float16)r[j]; a[8 + j] = (_Float16)r[8 + j]; }
    return a;
}
// B 32x16 (row-major B[k][n], stride ldb): lane n = lane&15; kb=(lane>=16)?16:0; elem j -> K=kb+j
__device__ inline v16h load_b_f16(const _Float16* B, int ldb, int lane) {
    int n = lane & 15; int kb = (lane & 16) ? 16 : 0;
    v16h b;
#pragma unroll
    for (int j = 0; j < 16; ++j) b[j] = B[(kb + j) * ldb + n];
    return b;
}
__device__ inline v16h load_b_f32(const float* B, int ldb, int lane) {
    int n = lane & 15; int kb = (lane & 16) ? 16 : 0;
    v16h b;
#pragma unroll
    for (int j = 0; j < 16; ++j) b[j] = (_Float16)B[(size_t)(kb + j) * ldb + n];
    return b;
}
// B = Kmat^T : B[k][n] = Kt[n*ldk + k] (contiguous in k per lane)
__device__ inline v16h load_bT_f16(const _Float16* Kt, int ldk, int lane) {
    int n = lane & 15; int kb = (lane & 16) ? 16 : 0;
    const _Float16* r = Kt + n * ldk + kb;
    v16h b;
#pragma unroll
    for (int j = 0; j < 16; ++j) b[j] = r[j];
    return b;
}

// ---- TDM: 1-D contiguous copy of nelem f16 elements global -> LDS ----
// D# per CDNA5 ISA 08_async_tensor: group0 {count=1, lds_addr, global_addr, type=2},
// group1 {data_size=2B, tensor_dim0=tile_dim0=nelem, dim1=1, dim0_stride=nelem}.
// This toolchain exposes the 6-arg builtin: (v4u, v8i, v4i, v4i, v8i, i32 cpol).
__device__ inline void tdm_copy_1d(const void* gsrc, const void* lds_dst, unsigned nelem) {
    unsigned long long ga = (unsigned long long)(uintptr_t)gsrc;
    unsigned lds_off = (unsigned)(uintptr_t)lds_dst;   // low 32 bits of LDS aperture addr
    v4u g0;
    g0[0] = 1u;                                         // count=1, user mode
    g0[1] = lds_off;                                    // lds_addr
    g0[2] = (unsigned)(ga & 0xFFFFFFFFu);               // global_addr[31:0]
    g0[3] = (unsigned)((ga >> 32) & 0x01FFFFFFu) | (2u << 30);  // addr[56:32], type=2
    v8i g1;
    g1[0] = (int)(1u << 16);                            // workgroup_mask=0, data_size=1 (2B)
    g1[1] = (int)((nelem & 0xFFFFu) << 16);             // tensor_dim0[15:0] in bits[63:48]
    g1[2] = (int)(((nelem >> 16) & 0xFFFFu) | (1u << 16)); // dim0[31:16], tensor_dim1=1
    g1[3] = (int)((nelem & 0xFFFFu) << 16);             // tile_dim0 = nelem (bits[127:112])
    g1[4] = 1;                                          // tile_dim1=1, tile_dim2=0
    g1[5] = (int)nelem;                                 // tensor_dim0_stride[31:0]
    g1[6] = 0;                                          // stride[47:32], dim1_stride lo
    g1[7] = 0;
    v4i z4 = {0, 0, 0, 0};
    v8i z8 = {0, 0, 0, 0, 0, 0, 0, 0};
    __builtin_amdgcn_tensor_load_to_lds(g0, g1, z4, z4, z8, 0);
}

// ---- attn2 column-sum reciprocals: s[j] = sum_m exp(-|j-m|/e) ----
__global__ void __launch_bounds__(256) k_colsum(float* inv_s) {
    int j = blockIdx.x * 256 + threadIdx.x;
    if (j >= Nn) return;
    float invE = __expf(-1.0f);
    float s = 0.f;
    for (int m = 0; m < Nn; ++m) s += __expf(-fabsf((float)(j - m)) * invE);
    inv_s[j] = 1.0f / s;
}

// ---- attn2[i][j] = exp(-|i-j|/e) * inv_s[j], stored f16 ----
__global__ void __launch_bounds__(256) k_attn2(const float* __restrict__ inv_s,
                                               _Float16* __restrict__ a2) {
    int idx = blockIdx.x * 256 + threadIdx.x;          // 0 .. 4M-1
    int i = idx >> 11, j = idx & (Nn - 1);
    float invE = __expf(-1.0f);
    a2[idx] = (_Float16)(__expf(-fabsf((float)(i - j)) * invE) * inv_s[j]);
}

// ---- GEMM1: qkvt = x @ W_qkv, scattered to head-major f16 q/k/v/t ----
__global__ void __launch_bounds__(256) k_qkv(const float* __restrict__ x,
                                             const float* __restrict__ W,
                                             _Float16* qh, _Float16* kh,
                                             _Float16* vh, _Float16* th) {
    int wid = blockIdx.x * 8 + WAVE;          // 32768 wave-tiles
    int mt = wid >> 7;                        // 0..255  (M = 4096)
    int nt = wid & 127;                       // 0..127  (Nout = 2048)
    int lane = LANE;
    const float* Ap = x + (size_t)(mt * 16) * DIMc;
    const float* Bp = W + nt * 16;
    v8f acc = {};
    for (int k0 = 0; k0 < DIMc; k0 += 32) {
        __builtin_prefetch(Bp + (size_t)(k0 + 32) * (4 * DIMc), 0, 1);
        v16h a = load_a_f32(Ap + k0, DIMc, lane);
        v16h b = load_b_f32(Bp + (size_t)k0 * (4 * DIMc), 4 * DIMc, lane);
        acc = wmma_f16(a, b, acc);
    }
    int col0 = nt * 16;
    int which = col0 >> 9;                    // 0=q 1=k 2=v 3=t
    int inner = col0 & 511;
    int hh = inner >> 6, dbase = inner & 63;
    _Float16* dst = which == 0 ? qh : which == 1 ? kh : which == 2 ? vh : th;
    int n = lane & 15, hi = lane >> 4;
#pragma unroll
    for (int r = 0; r < 8; ++r) {
        int m = mt * 16 + r + 8 * hi;
        int bb = m >> 11, nn = m & (Nn - 1);
        dst[((size_t)(bb * Hc + hh) * Nn + nn) * DHc + dbase + n] = (_Float16)acc[r];
    }
}

// ---- Flash attention with TDM-staged K/V chunks ----
// out1 = softmax(q_remap k^T / 8) v -> cat[..., h*128 + 0..63]
__global__ void __launch_bounds__(256) k_attn(const _Float16* __restrict__ qh,
                                              const _Float16* __restrict__ kh,
                                              const _Float16* __restrict__ vh,
                                              _Float16* __restrict__ cat) {
    __shared__ _Float16 Klds[CH * DHc];       // 16 KB
    __shared__ _Float16 Vlds[CH * DHc];       // 16 KB
    __shared__ _Float16 plds[8][16 * 32];     // 8 KB
    int bh = blockIdx.y;                      // 0..15 (whole block shares this head)
    int itile = blockIdx.x * 8 + WAVE;        // 0..127
    int lane = LANE;
    int n = lane & 15, hi = lane >> 4;
    const _Float16* Q = qh + (size_t)bh * Nn * DHc;
    const _Float16* Kp = kh + (size_t)bh * Nn * DHc;
    const _Float16* V = vh + (size_t)bh * Nn * DHc;

    // q fragments (dims 0..31 and 32..63), rows remapped: q_final[i] = q[i % 513]
    int qrow = (itile * 16 + n) % 513;
    v16h aq0, aq1;
    {
        int kb = hi ? 16 : 0;
        const _Float16* r0 = Q + (size_t)qrow * DHc;
#pragma unroll
        for (int j = 0; j < 8; ++j) {
            aq0[j] = r0[kb + j];           aq0[8 + j] = r0[kb + 8 + j];
            aq1[j] = r0[32 + kb + j];      aq1[8 + j] = r0[32 + kb + 8 + j];
        }
    }

    float mrun[8], lrun[8];
#pragma unroll
    for (int r = 0; r < 8; ++r) { mrun[r] = -1e30f; lrun[r] = 0.f; }
    v8f acc4[4] = {};
    _Float16* P = plds[WAVE];

    for (int c0 = 0; c0 < Nn; c0 += CH) {
        __syncthreads();                       // previous chunk fully consumed (WAR)
        if (WAVE == 0) {                       // TDM ignores EXEC: issue once per block
            tdm_copy_1d(Kp + (size_t)c0 * DHc, Klds, CH * DHc);
            tdm_copy_1d(V + (size_t)c0 * DHc, Vlds, CH * DHc);
            __builtin_amdgcn_s_wait_tensorcnt((short)0);
        }
        __syncthreads();                       // chunk visible to all waves

        for (int jr = 0; jr < CH; jr += 32) {
            v8f s0 = {}, s1 = {};
            s0 = wmma_f16(aq0, load_bT_f16(Klds + jr * DHc, DHc, lane), s0);
            s0 = wmma_f16(aq1, load_bT_f16(Klds + jr * DHc + 32, DHc, lane), s0);
            s1 = wmma_f16(aq0, load_bT_f16(Klds + (jr + 16) * DHc, DHc, lane), s1);
            s1 = wmma_f16(aq1, load_bT_f16(Klds + (jr + 16) * DHc + 32, DHc, lane), s1);

            float esc[8];
#pragma unroll
            for (int r = 0; r < 8; ++r) {
                float x0 = s0[r] * 0.125f, x1 = s1[r] * 0.125f;
                float tm = fmaxf(x0, x1);
#pragma unroll
                for (int off = 1; off < 16; off <<= 1) tm = fmaxf(tm, __shfl_xor(tm, off, 32));
                float mn = fmaxf(mrun[r], tm);
                esc[r] = __expf(mrun[r] - mn);
                float p0 = __expf(x0 - mn), p1 = __expf(x1 - mn);
                s0[r] = p0; s1[r] = p1;
                float rs = p0 + p1;
#pragma unroll
                for (int off = 1; off < 16; off <<= 1) rs += __shfl_xor(rs, off, 32);
                lrun[r] = lrun[r] * esc[r] + rs;
                mrun[r] = mn;
            }
            // C-layout -> A-layout transpose through LDS
#pragma unroll
            for (int r = 0; r < 8; ++r) {
                int row = r + 8 * hi;
                P[row * 32 + n]      = (_Float16)s0[r];
                P[row * 32 + 16 + n] = (_Float16)s1[r];
            }
            asm volatile("s_wait_dscnt 0" ::: "memory");
            v16h pf = load_a_f16(P, 32, lane);
#pragma unroll
            for (int f = 0; f < 4; ++f) {
#pragma unroll
                for (int r = 0; r < 8; ++r) acc4[f][r] *= esc[r];
                v16h bv = load_b_f16(Vlds + jr * DHc + f * 16, DHc, lane);
                acc4[f] = wmma_f16(pf, bv, acc4[f]);
            }
        }
    }

    int bb = bh >> 3, hh = bh & 7;
#pragma unroll
    for (int f = 0; f < 4; ++f)
#pragma unroll
        for (int r = 0; r < 8; ++r) {
            int i = itile * 16 + r + 8 * hi;
            cat[((size_t)(bb * Nn + i)) * CATC + hh * 128 + f * 16 + n] =
                (_Float16)(acc4[f][r] / lrun[r]);
        }
}

// ---- out2 = attn2 @ t -> cat[..., h*128 + 64..127] ----
__global__ void __launch_bounds__(256) k_out2(const _Float16* __restrict__ a2,
                                              const _Float16* __restrict__ th,
                                              _Float16* __restrict__ cat) {
    int wid = blockIdx.x * 8 + WAVE;          // 8192 wave-tiles
    int bh = wid >> 9;                        // 0..15
    int rem = wid & 511;
    int it = rem >> 2;                        // 0..127
    int dt = rem & 3;                         // 0..3
    int lane = LANE; int n = lane & 15, hi = lane >> 4;
    const _Float16* A = a2 + (size_t)(it * 16) * Nn;
    const _Float16* T = th + (size_t)bh * Nn * DHc + dt * 16;
    v8f acc = {};
    for (int k0 = 0; k0 < Nn; k0 += 32) {
        v16h a = load_a_f16(A + k0, Nn, lane);
        v16h b = load_b_f16(T + (size_t)k0 * DHc, DHc, lane);
        acc = wmma_f16(a, b, acc);
    }
    int bb = bh >> 3, hh = bh & 7;
#pragma unroll
    for (int r = 0; r < 8; ++r) {
        int i = it * 16 + r + 8 * hi;
        cat[((size_t)(bb * Nn + i)) * CATC + hh * 128 + 64 + dt * 16 + n] = (_Float16)acc[r];
    }
}

// ---- final projection: out = cat @ W_out + b_out (f32 output) ----
__global__ void __launch_bounds__(256) k_proj(const _Float16* __restrict__ cat,
                                              const float* __restrict__ W,
                                              const float* __restrict__ bias,
                                              float* __restrict__ out) {
    int wid = blockIdx.x * 8 + WAVE;          // 8192 wave-tiles
    int mt = wid >> 5;                        // 0..255 (M = 4096)
    int nt = wid & 31;                        // 0..31  (N = 512)
    int lane = LANE; int n = lane & 15, hi = lane >> 4;
    const _Float16* A = cat + (size_t)(mt * 16) * CATC;
    const float* Bp = W + nt * 16;
    v8f acc = {};
    for (int k0 = 0; k0 < CATC; k0 += 32) {
        __builtin_prefetch(Bp + (size_t)(k0 + 32) * DIMc, 0, 1);
        v16h a = load_a_f16(A + k0, CATC, lane);
        v16h b = load_b_f32(Bp + (size_t)k0 * DIMc, DIMc, lane);
        acc = wmma_f16(a, b, acc);
    }
    float bb = bias[nt * 16 + n];
#pragma unroll
    for (int r = 0; r < 8; ++r) {
        int m = mt * 16 + r + 8 * hi;
        out[(size_t)m * DIMc + nt * 16 + n] = acc[r] + bb;
    }
}

extern "C" void kernel_launch(void* const* d_in, const int* in_sizes, int n_in,
                              void* d_out, int out_size, void* d_ws, size_t ws_size,
                              hipStream_t stream) {
    (void)in_sizes; (void)n_in; (void)out_size; (void)ws_size;
    const float* x     = (const float*)d_in[0];
    const float* W_qkv = (const float*)d_in[1];
    const float* W_out = (const float*)d_in[2];
    const float* b_out = (const float*)d_in[3];
    float* out = (float*)d_out;

    const size_t HEAD_ELEMS = (size_t)Bc * Hc * Nn * DHc;     // 2,097,152
    char* ws = (char*)d_ws;
    _Float16* qh   = (_Float16*)ws;
    _Float16* kh   = qh + HEAD_ELEMS;
    _Float16* vh   = kh + HEAD_ELEMS;
    _Float16* th   = vh + HEAD_ELEMS;
    _Float16* a2   = th + HEAD_ELEMS;                         // Nn*Nn
    _Float16* cat  = a2 + (size_t)Nn * Nn;                    // Bc*Nn*CATC
    float*    invs = (float*)(cat + (size_t)Bc * Nn * CATC);

    k_colsum<<<Nn / 256, 256, 0, stream>>>(invs);
    k_attn2<<<(Nn * Nn) / 256, 256, 0, stream>>>(invs, a2);
    k_qkv<<<4096, 256, 0, stream>>>(x, W_qkv, qh, kh, vh, th);
    k_attn<<<dim3(16, Bc * Hc), 256, 0, stream>>>(qh, kh, vh, cat);
    k_out2<<<1024, 256, 0, stream>>>(a2, th, cat);
    k_proj<<<1024, 256, 0, stream>>>(cat, W_out, b_out, out);
}
